// Net_82197084110867
// MI455X (gfx1250) — matrix operand order repaired and checked by reading, not compiled, not run on hardware
//
#include <hip/hip_runtime.h>
#include <hip/hip_bf16.h>

typedef __bf16 bf16;
typedef __attribute__((ext_vector_type(16))) __bf16 v16bf;
typedef __attribute__((ext_vector_type(8)))  __bf16 v8bf;
typedef __attribute__((ext_vector_type(8)))  float  v8f;

#define NEG_SLOPE 0.2f

// ---------- helpers ----------
__device__ __forceinline__ unsigned fkey(float f) {
    unsigned b = __float_as_uint(f);
    return (b & 0x80000000u) ? ~b : (b | 0x80000000u);   // order-preserving map
}
__device__ __forceinline__ float unkey(unsigned k) {
    unsigned b = (k & 0x80000000u) ? (k & 0x7fffffffu) : ~k;
    return __uint_as_float(b);
}

// ---------- f32 -> bf16 with zero padding (+ optional transpose) ----------
__global__ void pad_convert_kernel(const float* __restrict__ src, bf16* __restrict__ dst,
                                   int srcRows, int srcCols, int dstRows, int dstCols,
                                   int transpose) {
    long idx = (long)blockIdx.x * blockDim.x + threadIdx.x;
    long total = (long)dstRows * dstCols;
    if (idx >= total) return;
    int r = (int)(idx / dstCols), c = (int)(idx % dstCols);
    float v = 0.f;
    if (transpose) { if (c < srcRows && r < srcCols) v = src[(long)c * srcCols + r]; }
    else           { if (r < srcRows && c < srcCols) v = src[(long)r * srcCols + c]; }
    dst[idx] = (bf16)v;
}

// ---------- WMMA GEMM: C(MxN,f32) = A(Mp x Kp, bf16) * B(Kp x Np, bf16) [+bias][relu] ----
// A rows padded to mult-64, Kp mult-32, Np (ldb) mult-64; pads are zero -> no inner checks.
// Block tile 64x64, 8 wave32s; each wave computes one 16x16 M-tile x two 16-wide N-tiles
// (A fragment reused across 2 WMMAs).
#define BM 64
#define BN 64
#define BK 32
#define AS_LD 40
#define BS_LD 40

__global__ __launch_bounds__(256)
void wmma_gemm_bf16(const bf16* __restrict__ A, const bf16* __restrict__ B,
                    const float* __restrict__ bias, float* __restrict__ C,
                    int M, int N, int Kp, int ldb, int relu) {
    __shared__ alignas(16) bf16 As[BM * AS_LD];          // row-major [m][k]
    __shared__ alignas(16) bf16 Bs[BN * BS_LD];          // transposed [n][k]
    const int tid  = threadIdx.x;
    const int wave = tid >> 5;
    const int lane = tid & 31;
    const int wm = wave & 3;          // M sub-tile 0..3
    const int wn = wave >> 2;         // N half 0..1 (each covers 32 cols = 2 WMMA tiles)
    const int g  = lane >> 4;         // lane half (ISA fragment layout)
    const int nl = lane & 15;
    const int m0 = blockIdx.x * BM;
    const int n0 = blockIdx.y * BN;

    const int arow  = tid >> 2;        // 0..63
    const int akseg = (tid & 3) * 8;   // 0,8,16,24
    const int bk    = tid >> 3;        // 0..31
    const int bnseg = (tid & 7) * 8;   // 0..56

    v8f acc0 = {0.f, 0.f, 0.f, 0.f, 0.f, 0.f, 0.f, 0.f};
    v8f acc1 = {0.f, 0.f, 0.f, 0.f, 0.f, 0.f, 0.f, 0.f};

    for (int k0 = 0; k0 < Kp; k0 += BK) {
        // A tile: aligned 16B global load -> aligned 16B LDS store
        v8bf av = *reinterpret_cast<const v8bf*>(A + (size_t)(m0 + arow) * Kp + k0 + akseg);
        *reinterpret_cast<v8bf*>(&As[arow * AS_LD + akseg]) = av;
        // B tile: aligned 16B global load, transpose into LDS [n][k]
        v8bf bv = *reinterpret_cast<const v8bf*>(B + (size_t)(k0 + bk) * ldb + n0 + bnseg);
#pragma unroll
        for (int i = 0; i < 8; ++i) Bs[(bnseg + i) * BS_LD + bk] = bv[i];
        __syncthreads();

        // A fragment: lanes 0-15 hold K 0..15, lanes 16-31 hold K 16..31
        const int ml = wm * 16 + nl;
        v8bf a0 = *reinterpret_cast<const v8bf*>(&As[ml * AS_LD + g * 16]);
        v8bf a1 = *reinterpret_cast<const v8bf*>(&As[ml * AS_LD + g * 16 + 8]);
        v16bf af;
#pragma unroll
        for (int i = 0; i < 8; ++i) { af[i] = a0[i]; af[i + 8] = a1[i]; }

        // B fragments (two 16-col tiles): lanes 0-15 hold K 0..15, lanes 16-31 K 16..31
        const int nn0 = wn * 32 + nl;
        v8bf b00 = *reinterpret_cast<const v8bf*>(&Bs[nn0 * BS_LD + g * 16]);
        v8bf b01 = *reinterpret_cast<const v8bf*>(&Bs[nn0 * BS_LD + g * 16 + 8]);
        const int nn1 = nn0 + 16;
        v8bf b10 = *reinterpret_cast<const v8bf*>(&Bs[nn1 * BS_LD + g * 16]);
        v8bf b11 = *reinterpret_cast<const v8bf*>(&Bs[nn1 * BS_LD + g * 16 + 8]);
        v16bf bf0, bf1;
#pragma unroll
        for (int i = 0; i < 8; ++i) { bf0[i] = b00[i]; bf0[i + 8] = b01[i];
                                      bf1[i] = b10[i]; bf1[i + 8] = b11[i]; }

        acc0 = __builtin_amdgcn_wmma_f32_16x16x32_bf16(false, af, false, bf0,
                                                       (short)0, acc0, false, false);
        acc1 = __builtin_amdgcn_wmma_f32_16x16x32_bf16(false, af, false, bf1,
                                                       (short)0, acc1, false, false);
        __syncthreads();
    }

    const int gmBase = m0 + wm * 16 + g * 8;
    const int gn0 = n0 + wn * 32 + nl;
    const int gn1 = gn0 + 16;
    if (gn0 < N) {
        float bb = bias ? bias[gn0] : 0.f;
#pragma unroll
        for (int r = 0; r < 8; ++r) {
            int gm = gmBase + r;
            if (gm < M) {
                float v = acc0[r] + bb;
                if (relu) v = fmaxf(v, 0.f);
                C[(size_t)gm * N + gn0] = v;
            }
        }
    }
    if (gn1 < N) {
        float bb = bias ? bias[gn1] : 0.f;
#pragma unroll
        for (int r = 0; r < 8; ++r) {
            int gm = gmBase + r;
            if (gm < M) {
                float v = acc1[r] + bb;
                if (relu) v = fmaxf(v, 0.f);
                C[(size_t)gm * N + gn1] = v;
            }
        }
    }
}

// ---------- hypergraph attention primitives ----------
__global__ void deg_kernel(const int* __restrict__ hei, float* __restrict__ deg, int E) {
    int e = blockIdx.x * blockDim.x + threadIdx.x;
    if (e >= E) return;
    atomicAdd(&deg[hei[e]], 1.0f);
}
__global__ void dinv_kernel(const float* __restrict__ deg, float* __restrict__ dinv, int Nn) {
    int i = blockIdx.x * blockDim.x + threadIdx.x;
    if (i >= Nn) return;
    float d = deg[i];
    dinv[i] = d > 0.f ? 1.0f / d : 0.f;
}
__global__ void alpha_kernel(const float* __restrict__ Xp, const int* __restrict__ hei,
                             const float* __restrict__ att, float* __restrict__ alpha,
                             unsigned* __restrict__ amax, int E, int H, int O) {
    int idx = blockIdx.x * blockDim.x + threadIdx.x;
    if (idx >= E * H) return;
    int e = idx / H, h = idx - e * H;
    int r = hei[e], c = hei[E + e];
    int HO = H * O;
    const float* xi = Xp + (size_t)r * HO + h * O;
    const float* xj = Xp + (size_t)c * HO + h * O;
    const float* at = att + (size_t)h * 2 * O;
    float a = 0.f;
    for (int t = 0; t < O; ++t) a += xi[t] * at[t] + xj[t] * at[O + t];
    a = (a >= 0.f) ? a : NEG_SLOPE * a;                  // leaky relu
    alpha[idx] = a;
    atomicMax(&amax[(size_t)r * H + h], fkey(a));        // segment max
}
__global__ void expsum_kernel(const int* __restrict__ hei, float* __restrict__ alpha,
                              const unsigned* __restrict__ amax, float* __restrict__ asum,
                              int E, int H) {
    int idx = blockIdx.x * blockDim.x + threadIdx.x;
    if (idx >= E * H) return;
    int e = idx / H, h = idx - e * H;
    int r = hei[e];
    float v = __expf(alpha[idx] - unkey(amax[(size_t)r * H + h]));
    alpha[idx] = v;
    atomicAdd(&asum[(size_t)r * H + h], v);
}
__global__ void norm_kernel(const int* __restrict__ hei, float* __restrict__ alpha,
                            const float* __restrict__ asum, int E, int H) {
    int idx = blockIdx.x * blockDim.x + threadIdx.x;
    if (idx >= E * H) return;
    int e = idx / H, h = idx - e * H;
    int r = hei[e];
    alpha[idx] = alpha[idx] / (asum[(size_t)r * H + h] + 1e-16f);
}
__global__ void scatter1_kernel(const float* __restrict__ Xp, const int* __restrict__ hei,
                                const float* __restrict__ alpha, const float* __restrict__ dinv,
                                float* __restrict__ out1, int E, int H, int O) {
    long idx = (long)blockIdx.x * blockDim.x + threadIdx.x;
    int HO = H * O;
    if (idx >= (long)E * HO) return;
    int e = (int)(idx / HO), j = (int)(idx - (long)e * HO);
    int h = j / O;
    int r = hei[e], c = hei[E + e];
    float v = dinv[c] * Xp[(size_t)r * HO + j] * alpha[(size_t)e * H + h];
    atomicAdd(&out1[(size_t)c * HO + j], v);
}
__global__ void scatter2_kernel(const float* __restrict__ out1, const int* __restrict__ hei,
                                const float* __restrict__ alpha, const float* __restrict__ dinv,
                                float* __restrict__ out2, int E, int H, int O) {
    long idx = (long)blockIdx.x * blockDim.x + threadIdx.x;
    int HO = H * O;
    if (idx >= (long)E * HO) return;
    int e = (int)(idx / HO), j = (int)(idx - (long)e * HO);
    int h = j / O;
    int r = hei[e], c = hei[E + e];
    float v = dinv[r] * out1[(size_t)c * HO + j] * alpha[(size_t)e * H + h];
    atomicAdd(&out2[(size_t)r * HO + j], v);
}

// ---------- row softmax: one wave32 per row ----------
__global__ void softmax_rows(float* __restrict__ x, int M, int N) {
    int row = blockIdx.x * (blockDim.x >> 5) + (threadIdx.x >> 5);
    int lane = threadIdx.x & 31;
    if (row >= M) return;
    float* p = x + (size_t)row * N;
    float m = -3.4e38f;
    for (int j = lane; j < N; j += 32) m = fmaxf(m, p[j]);
#pragma unroll
    for (int s = 16; s > 0; s >>= 1) m = fmaxf(m, __shfl_xor(m, s, 32));
    float sum = 0.f;
    for (int j = lane; j < N; j += 32) sum += __expf(p[j] - m);
#pragma unroll
    for (int s = 16; s > 0; s >>= 1) sum += __shfl_xor(sum, s, 32);
    float inv = 1.0f / sum;
    for (int j = lane; j < N; j += 32) p[j] = __expf(p[j] - m) * inv;
}

// ---------- host-side attention stage ----------
static void run_attention(const float* Xp, const int* hei, const float* att,
                          float* alpha, unsigned* amax, float* asum, const float* dinv,
                          float* o1, float* o2, int Nn, int E, int HE, int H, int O,
                          hipStream_t stream) {
    int HO = H * O;
    hipMemsetAsync(amax, 0, (size_t)Nn * H * 4, stream);   // 0 == mapped -inf
    hipMemsetAsync(asum, 0, (size_t)Nn * H * 4, stream);
    hipMemsetAsync(o1, 0, (size_t)HE * HO * 4, stream);
    hipMemsetAsync(o2, 0, (size_t)Nn * HO * 4, stream);
    int EH = E * H;
    alpha_kernel<<<(EH + 255) / 256, 256, 0, stream>>>(Xp, hei, att, alpha, amax, E, H, O);
    expsum_kernel<<<(EH + 255) / 256, 256, 0, stream>>>(hei, alpha, amax, asum, E, H);
    norm_kernel<<<(EH + 255) / 256, 256, 0, stream>>>(hei, alpha, asum, E, H);
    long EHO = (long)E * HO;
    scatter1_kernel<<<(unsigned)((EHO + 255) / 256), 256, 0, stream>>>(Xp, hei, alpha, dinv, o1, E, H, O);
    scatter2_kernel<<<(unsigned)((EHO + 255) / 256), 256, 0, stream>>>(o1, hei, alpha, dinv, o2, E, H, O);
}

extern "C" void kernel_launch(void* const* d_in, const int* in_sizes, int n_in,
                              void* d_out, int out_size, void* d_ws, size_t ws_size,
                              hipStream_t stream) {
    (void)in_sizes; (void)n_in; (void)out_size; (void)ws_size;
    const float* x    = (const float*)d_in[0];
    const int*   hei  = (const int*)d_in[1];
    const float* w1   = (const float*)d_in[2];
    const float* att1 = (const float*)d_in[3];
    const float* fc1w = (const float*)d_in[4];
    const float* fc1b = (const float*)d_in[5];
    const float* w2   = (const float*)d_in[6];
    const float* att2 = (const float*)d_in[7];
    const float* fc2w = (const float*)d_in[8];
    const float* fc2b = (const float*)d_in[9];
    float* out = (float*)d_out;

    const int N = 50000, E = 100000, HE = 10000, H = 6;
    const int O1 = 16, O2 = 223, HO1 = 96, HO2 = 1338;
    const int Mp = 50048;                                  // 50000 -> mult 64
    const int Kp0 = 512, NpW1 = 128;                       // layer1 proj (96 -> 128)
    const int KpFc1 = 96, NpFc1 = 64;                      // fc1 (16 -> 64)
    const int KpH = 32, NpW2 = 1344;                       // layer2 proj (16->32, 1338->1344)
    const int KpFc2 = 1344, NpFc2 = 256;                   // fc2 (223 -> 256)

    char* base = (char*)d_ws; size_t off = 0;
    auto alloc = [&](size_t bytes) -> void* {
        void* p = base + off; off = (off + bytes + 255) & ~(size_t)255; return p;
    };

    // persistent region
    float*    alpha = (float*)alloc((size_t)E * H * 4);
    unsigned* amax  = (unsigned*)alloc((size_t)N * H * 4);
    float*    asum  = (float*)alloc((size_t)N * H * 4);
    float*    deg   = (float*)alloc((size_t)N * 4);
    float*    dinv  = (float*)alloc((size_t)N * 4);
    float*    hbuf  = (float*)alloc((size_t)N * O1 * 4);   // fc1 output, persists to L2
    size_t arena = off;

    // layer-1 arena
    bf16*  Xbf    = (bf16*)alloc((size_t)Mp * Kp0 * 2);
    bf16*  W1bf   = (bf16*)alloc((size_t)Kp0 * NpW1 * 2);
    float* Xp1    = (float*)alloc((size_t)N * HO1 * 4);
    float* o1_1   = (float*)alloc((size_t)HE * HO1 * 4);
    float* o2_1   = (float*)alloc((size_t)N * HO1 * 4);
    bf16*  o2_1bf = (bf16*)alloc((size_t)Mp * KpFc1 * 2);
    bf16*  fc1T   = (bf16*)alloc((size_t)KpFc1 * NpFc1 * 2);

    // layer-2 arena (aliases layer-1; stream order serializes reuse)
    off = arena;
    bf16*  hbf    = (bf16*)alloc((size_t)Mp * KpH * 2);
    bf16*  W2bf   = (bf16*)alloc((size_t)KpH * NpW2 * 2);
    float* Xp2    = (float*)alloc((size_t)N * HO2 * 4);
    float* o1_2   = (float*)alloc((size_t)HE * HO2 * 4);
    float* o2_2   = (float*)alloc((size_t)N * HO2 * 4);
    bf16*  o2_2bf = (bf16*)alloc((size_t)Mp * KpFc2 * 2);
    bf16*  fc2T   = (bf16*)alloc((size_t)KpFc2 * NpFc2 * 2);

    auto gblk = [](long n) { return (unsigned)((n + 255) / 256); };

    // degrees (row-based, same for both layers)
    hipMemsetAsync(deg, 0, (size_t)N * 4, stream);
    deg_kernel<<<gblk(E), 256, 0, stream>>>(hei, deg, E);
    dinv_kernel<<<gblk(N), 256, 0, stream>>>(deg, dinv, N);

    // ---------------- layer 1 ----------------
    pad_convert_kernel<<<gblk((long)Mp * Kp0), 256, 0, stream>>>(x, Xbf, N, 512, Mp, Kp0, 0);
    pad_convert_kernel<<<gblk((long)Kp0 * NpW1), 256, 0, stream>>>(w1, W1bf, 512, 96, Kp0, NpW1, 0);
    wmma_gemm_bf16<<<dim3(Mp / 64, (HO1 + 63) / 64), 256, 0, stream>>>(
        Xbf, W1bf, nullptr, Xp1, N, HO1, Kp0, NpW1, 0);

    run_attention(Xp1, hei, att1, alpha, amax, asum, dinv, o1_1, o2_1, N, E, HE, H, O1, stream);

    pad_convert_kernel<<<gblk((long)Mp * KpFc1), 256, 0, stream>>>(o2_1, o2_1bf, N, HO1, Mp, KpFc1, 0);
    pad_convert_kernel<<<gblk((long)KpFc1 * NpFc1), 256, 0, stream>>>(fc1w, fc1T, 16, 96, KpFc1, NpFc1, 1);
    wmma_gemm_bf16<<<dim3(Mp / 64, 1), 256, 0, stream>>>(
        o2_1bf, fc1T, fc1b, hbuf, N, O1, KpFc1, NpFc1, 1);   // + ReLU

    // ---------------- layer 2 ----------------
    pad_convert_kernel<<<gblk((long)Mp * KpH), 256, 0, stream>>>(hbuf, hbf, N, 16, Mp, KpH, 0);
    pad_convert_kernel<<<gblk((long)KpH * NpW2), 256, 0, stream>>>(w2, W2bf, 16, 1338, KpH, NpW2, 0);
    wmma_gemm_bf16<<<dim3(Mp / 64, (HO2 + 63) / 64), 256, 0, stream>>>(
        hbf, W2bf, nullptr, Xp2, N, HO2, KpH, NpW2, 0);

    run_attention(Xp2, hei, att2, alpha, amax, asum, dinv, o1_2, o2_2, N, E, HE, H, O2, stream);

    pad_convert_kernel<<<gblk((long)Mp * KpFc2), 256, 0, stream>>>(o2_2, o2_2bf, N, HO2, Mp, KpFc2, 0);
    pad_convert_kernel<<<gblk((long)KpFc2 * NpFc2), 256, 0, stream>>>(fc2w, fc2T, 223, 1338, KpFc2, NpFc2, 1);
    wmma_gemm_bf16<<<dim3(Mp / 64, (223 + 63) / 64), 256, 0, stream>>>(
        o2_2bf, fc2T, fc2b, out, N, 223, KpFc2, NpFc2, 0);

    softmax_rows<<<(N + 7) / 8, 256, 0, stream>>>(out, N, 223);
}
